// GroupedQueryAttention_85323820303101
// MI455X (gfx1250) — matrix operand order repaired and checked by reading, not compiled, not run on hardware
//
#include <hip/hip_runtime.h>

#define DEV_INLINE __device__ __forceinline__

typedef __attribute__((ext_vector_type(16))) __bf16 v16bf;
typedef __attribute__((ext_vector_type(8)))  float  v8f;
typedef __attribute__((ext_vector_type(8)))  __bf16 bf16x8;
typedef __attribute__((ext_vector_type(4)))  __bf16 bf16x4;

// problem dims
constexpr int Bb   = 2;
constexpr int TQ   = 1024;
constexpr int DIN  = 4096;
constexpr int NH   = 32;
constexpr int NG   = 8;
constexpr int HD   = 128;
constexpr int SP   = 1024;
constexpr int STOT = 2048;

// ---------------------------------------------------------------------------
// WMMA fragment helpers (bf16, 16x16x32)
// ---------------------------------------------------------------------------
DEV_INLINE v16bf ldfragA(const __bf16* base, int ld) {
  const int lane = threadIdx.x & 31;
  const __bf16* p = base + (size_t)(lane & 15) * ld + ((lane >> 4) << 3);
  bf16x8 lo = *(const bf16x8*)p;
  bf16x8 hi = *(const bf16x8*)(p + 16);
  return __builtin_shufflevector(lo, hi, 0,1,2,3,4,5,6,7,8,9,10,11,12,13,14,15);
}
DEV_INLINE v16bf ldfragB(const __bf16* base, int ld) {
  const int lane = threadIdx.x & 31;
  const __bf16* p = base + (size_t)(lane & 15) * ld + ((lane >> 4) << 3);
  bf16x8 lo = *(const bf16x8*)p;
  bf16x8 hi = *(const bf16x8*)(p + 16);
  return __builtin_shufflevector(lo, hi, 0,1,2,3,4,5,6,7,8,9,10,11,12,13,14,15);
}
DEV_INLINE v8f wmma_bf16(v16bf a, v16bf b, v8f c) {
  return __builtin_amdgcn_wmma_f32_16x16x32_bf16(false, a, false, b, (short)0, c,
                                                 false, false);
}

// ---------------------------------------------------------------------------
// Tensor Data Mover helpers (2-D tile load with LDS row padding)
// ---------------------------------------------------------------------------
#if __has_builtin(__builtin_amdgcn_tensor_load_to_lds)
#define HAVE_TDM 1
typedef __attribute__((ext_vector_type(4))) unsigned int u32x4;
typedef __attribute__((ext_vector_type(4))) int          i32x4;
typedef __attribute__((ext_vector_type(8))) int          i32x8;

DEV_INLINE void tdm_wait0() {
#if __has_builtin(__builtin_amdgcn_s_wait_tensorcnt)
  __builtin_amdgcn_s_wait_tensorcnt(0);
#else
  asm volatile("s_wait_tensorcnt 0x0" ::: "memory");
#endif
}

// Load a tile_d1 x tile_d0 tile of 2-byte elements into LDS.
// tensor rows are stride_d0 elements apart; LDS rows get padding DWORDs
// after every 2^pad_interval * 8 bytes.
DEV_INLINE void tdm_load_2d(unsigned lds_off, const void* gaddr,
                            unsigned tensor_d0, unsigned tensor_d1,
                            unsigned tile_d0, unsigned tile_d1,
                            unsigned stride_d0,
                            unsigned pad_interval, unsigned pad_amount_m1) {
  unsigned long long ga = (unsigned long long)gaddr;
  u32x4 g0;
  g0.x = 1u;                                    // count=1 (user descriptor)
  g0.y = lds_off;                               // lds_addr [63:32]
  g0.z = (unsigned)(ga & 0xffffffffu);          // global_addr [95:64]
  g0.w = (unsigned)((ga >> 32) & 0x1ffffffu) | (2u << 30);  // addr hi + type=2
  i32x8 g1;
  g1[0] = (int)((1u << 16)                      // data_size = 2 bytes
                | (1u << 20)                    // pad_enable
                | (pad_interval << 22) | (pad_amount_m1 << 25));
  g1[1] = (int)(tensor_d0 << 16);               // tensor_dim0 lo16 @ [63:48]
  g1[2] = (int)((tensor_d0 >> 16) | (tensor_d1 << 16));
  g1[3] = (int)((tensor_d1 >> 16) | (tile_d0 << 16));
  g1[4] = (int)(tile_d1 & 0xffffu);             // tile_dim1, tile_dim2=0
  g1[5] = (int)stride_d0;                       // dim0_stride lo32
  g1[6] = (int)(stride_d0 >> 31 >> 1);          // dim0_stride hi16
  g1[7] = 0;
  i32x4 z4 = {0, 0, 0, 0};
  i32x8 z8 = {0, 0, 0, 0, 0, 0, 0, 0};
  __builtin_amdgcn_tensor_load_to_lds(g0, g1, z4, z4, z8, 0);
}
#endif

// ---------------------------------------------------------------------------
// f32 -> bf16 cast (4 elems / thread)
// ---------------------------------------------------------------------------
__global__ void cast_bf16_kernel(const float* __restrict__ in,
                                 __bf16* __restrict__ out, int n4) {
  int i = blockIdx.x * blockDim.x + threadIdx.x;
  if (i >= n4) return;
  float4 v = ((const float4*)in)[i];
  bf16x4 o;
  o.x = (__bf16)v.x; o.y = (__bf16)v.y; o.z = (__bf16)v.z; o.w = (__bf16)v.w;
  ((bf16x4*)out)[i] = o;
}

// ---------------------------------------------------------------------------
// C[M,N](f32) = A[M,K](bf16 rm) @ B[N,K](bf16 rm)^T
// 256 thr (8 waves), tile 128x128, BK=32; waves 4(M) x 2(N) = 32x64 each.
// Software-pipelined: global->reg prefetch of tile t+1 overlaps WMMA of t.
// ---------------------------------------------------------------------------
template <int N, int K>
__global__ __launch_bounds__(256) void gemm_bf16_nt_kernel(
    const __bf16* __restrict__ A, const __bf16* __restrict__ B,
    float* __restrict__ C) {
  __shared__ __bf16 sA[128][40];   // 32 + 8 pad
  __shared__ __bf16 sB[128][40];

  const int tid  = threadIdx.x;
  const int lane = tid & 31;
  const int wave = tid >> 5;
  const int wm = wave >> 1;        // 0..3
  const int wn = wave & 1;         // 0..1
  const int m0 = blockIdx.y * 128;
  const int n0 = blockIdx.x * 128;

  const int ldRow  = tid >> 1;     // 0..127
  const int ldHalf = (tid & 1) * 16;

  const __bf16* pa = A + (size_t)(m0 + ldRow) * K + ldHalf;
  const __bf16* pb = B + (size_t)(n0 + ldRow) * K + ldHalf;

  v8f acc[2][4];
#pragma unroll
  for (int i = 0; i < 2; ++i)
#pragma unroll
    for (int j = 0; j < 4; ++j) acc[i][j] = v8f{};

  // prologue: fetch tile 0 into registers
  bf16x8 ra0 = *(const bf16x8*)pa;
  bf16x8 ra1 = *(const bf16x8*)(pa + 8);
  bf16x8 rb0 = *(const bf16x8*)pb;
  bf16x8 rb1 = *(const bf16x8*)(pb + 8);

  constexpr int NT = K / 32;
  for (int t = 0; t < NT; ++t) {
    __syncthreads();               // prior tile's LDS reads complete
    *(bf16x8*)&sA[ldRow][ldHalf]     = ra0;
    *(bf16x8*)&sA[ldRow][ldHalf + 8] = ra1;
    *(bf16x8*)&sB[ldRow][ldHalf]     = rb0;
    *(bf16x8*)&sB[ldRow][ldHalf + 8] = rb1;
    __syncthreads();
    if (t + 1 < NT) {              // prefetch next tile (overlaps WMMA)
      const __bf16* na = pa + (size_t)(t + 1) * 32;
      const __bf16* nb = pb + (size_t)(t + 1) * 32;
      ra0 = *(const bf16x8*)na; ra1 = *(const bf16x8*)(na + 8);
      rb0 = *(const bf16x8*)nb; rb1 = *(const bf16x8*)(nb + 8);
    }
    v16bf af[2], bf[4];
#pragma unroll
    for (int i = 0; i < 2; ++i) af[i] = ldfragA(&sA[wm * 32 + i * 16][0], 40);
#pragma unroll
    for (int j = 0; j < 4; ++j) bf[j] = ldfragB(&sB[wn * 64 + j * 16][0], 40);
#pragma unroll
    for (int i = 0; i < 2; ++i)
#pragma unroll
      for (int j = 0; j < 4; ++j) acc[i][j] = wmma_bf16(af[i], bf[j], acc[i][j]);
  }

  const int halfadd = (lane & 16) ? 8 : 0;
  const int fcol    = lane & 15;
#pragma unroll
  for (int i = 0; i < 2; ++i)
#pragma unroll
    for (int j = 0; j < 4; ++j) {
      float* cp = C + (size_t)(m0 + wm * 32 + i * 16 + halfadd) * N +
                  (n0 + wn * 64 + j * 16 + fcol);
#pragma unroll
      for (int r = 0; r < 8; ++r) cp[r * N] = acc[i][j][r];  // const offsets
    }
}

// ---------------------------------------------------------------------------
// RMSNorm + RoPE for Q. One wave per (b,t,h). qraw: [B*T, NH*HD] f32.
// ---------------------------------------------------------------------------
__global__ void postq_kernel(const float* __restrict__ qraw,
                             const float* __restrict__ gw,
                             const float* __restrict__ cosT,
                             const float* __restrict__ sinT,
                             __bf16* __restrict__ qb) {
  int w    = (blockIdx.x * blockDim.x + threadIdx.x) >> 5;
  int lane = threadIdx.x & 31;
  int h  = w & (NH - 1);
  int bt = w >> 5;
  int b  = bt >> 10;
  int t  = bt & (TQ - 1);
  const float* src = qraw + (size_t)bt * (NH * HD) + h * HD;
  float x0 = src[lane], x1 = src[lane + 32], x2 = src[lane + 64], x3 = src[lane + 96];
  float ss = x0 * x0 + x1 * x1 + x2 * x2 + x3 * x3;
#pragma unroll
  for (int m = 1; m <= 16; m <<= 1) ss += __shfl_xor(ss, m, 32);
  float rs = rsqrtf(ss * (1.0f / 128.0f) + 1e-6f);
  x0 *= rs * gw[lane]; x1 *= rs * gw[lane + 32];
  x2 *= rs * gw[lane + 64]; x3 *= rs * gw[lane + 96];
  int pos = SP + t;
  const float* cp = cosT + (size_t)pos * HD;
  const float* sp = sinT + (size_t)pos * HD;
  float o0 = x0 * cp[lane]      - x2 * sp[lane];
  float o1 = x1 * cp[lane + 32] - x3 * sp[lane + 32];
  float o2 = x2 * cp[lane + 64] + x0 * sp[lane + 64];
  float o3 = x3 * cp[lane + 96] + x1 * sp[lane + 96];
  __bf16* dst = qb + (((size_t)(b * NH + h)) * TQ + t) * HD;
  dst[lane] = (__bf16)o0; dst[lane + 32] = (__bf16)o1;
  dst[lane + 64] = (__bf16)o2; dst[lane + 96] = (__bf16)o3;
}

// RMSNorm + RoPE for K. kvraw row stride 2048 (k in cols 0..1023).
__global__ void postk_kernel(const float* __restrict__ kvraw,
                             const float* __restrict__ gw,
                             const float* __restrict__ cosT,
                             const float* __restrict__ sinT,
                             float* __restrict__ kall,
                             __bf16* __restrict__ kbb) {
  int w    = (blockIdx.x * blockDim.x + threadIdx.x) >> 5;
  int lane = threadIdx.x & 31;
  int g  = w & (NG - 1);
  int bt = w >> 3;
  int b  = bt >> 10;
  int t  = bt & (TQ - 1);
  const float* src = kvraw + (size_t)bt * 2048 + g * HD;
  float x0 = src[lane], x1 = src[lane + 32], x2 = src[lane + 64], x3 = src[lane + 96];
  float ss = x0 * x0 + x1 * x1 + x2 * x2 + x3 * x3;
#pragma unroll
  for (int m = 1; m <= 16; m <<= 1) ss += __shfl_xor(ss, m, 32);
  float rs = rsqrtf(ss * (1.0f / 128.0f) + 1e-6f);
  x0 *= rs * gw[lane]; x1 *= rs * gw[lane + 32];
  x2 *= rs * gw[lane + 64]; x3 *= rs * gw[lane + 96];
  int pos = SP + t;
  const float* cp = cosT + (size_t)pos * HD;
  const float* sp = sinT + (size_t)pos * HD;
  float o0 = x0 * cp[lane]      - x2 * sp[lane];
  float o1 = x1 * cp[lane + 32] - x3 * sp[lane + 32];
  float o2 = x2 * cp[lane + 64] + x0 * sp[lane + 64];
  float o3 = x3 * cp[lane + 96] + x1 * sp[lane + 96];
  size_t o = (((size_t)(b * NG + g)) * STOT + (SP + t)) * HD;
  kall[o + lane] = o0; kall[o + lane + 32] = o1;
  kall[o + lane + 64] = o2; kall[o + lane + 96] = o3;
  kbb[o + lane] = (__bf16)o0; kbb[o + lane + 32] = (__bf16)o1;
  kbb[o + lane + 64] = (__bf16)o2; kbb[o + lane + 96] = (__bf16)o3;
}

// V scatter: v_all (f32) and transposed bf16 [B,G,HD,S]. v in cols 1024..2047.
__global__ void postv_kernel(const float* __restrict__ kvraw,
                             float* __restrict__ vall,
                             __bf16* __restrict__ vbt) {
  int w    = (blockIdx.x * blockDim.x + threadIdx.x) >> 5;
  int lane = threadIdx.x & 31;
  int g  = w & (NG - 1);
  int bt = w >> 3;
  int b  = bt >> 10;
  int t  = bt & (TQ - 1);
  const float* src = kvraw + (size_t)bt * 2048 + 1024 + g * HD;
  size_t od = (((size_t)(b * NG + g)) * STOT + (SP + t)) * HD;
  size_t ot = ((size_t)(b * NG + g)) * HD * STOT + (SP + t);
#pragma unroll
  for (int i = 0; i < 4; ++i) {
    int d = lane + 32 * i;
    float v = src[d];
    vall[od + d] = v;
    vbt[ot + (size_t)d * STOT] = (__bf16)v;
  }
}

// prev_k/prev_v -> k_all/v_all (f32) + kb bf16 + vbt bf16 (transposed)
__global__ void copyprev_kernel(const float* __restrict__ pk,
                                const float* __restrict__ pv,
                                float* __restrict__ kall,
                                float* __restrict__ vall,
                                __bf16* __restrict__ kbb,
                                __bf16* __restrict__ vbt) {
  int idx = blockIdx.x * blockDim.x + threadIdx.x;
  if (idx >= Bb * NG * SP * HD) return;
  int d = idx & (HD - 1);
  int s = (idx >> 7) & (SP - 1);
  int g = (idx >> 17) & (NG - 1);
  int b = idx >> 20;
  float kv = pk[idx], vv = pv[idx];
  size_t o = (((size_t)(b * NG + g)) * STOT + s) * HD + d;
  kall[o] = kv; kbb[o] = (__bf16)kv; vall[o] = vv;
  vbt[((size_t)(b * NG + g)) * HD * STOT + (size_t)d * STOT + s] = (__bf16)vv;
}

// ---------------------------------------------------------------------------
// Flash attention. 128 thr (4 waves); each wave = 16 q rows x HD=128.
// grid: (TQ/64, B*NH). K / V^T chunks staged into LDS via the Tensor Data
// Mover (wave 0 issues the descriptors; padding fields reproduce the +8-elem
// LDS row padding), fallback to manual staging when TDM builtin is absent.
// ---------------------------------------------------------------------------
__global__ __launch_bounds__(128) void attn_kernel(
    const __bf16* __restrict__ qb, const __bf16* __restrict__ kb,
    const __bf16* __restrict__ vbt, __bf16* __restrict__ ctxb) {
  __shared__ __bf16 sK[64][136];      // K chunk   [s, d], 8-elem row pad
  __shared__ __bf16 sVT[128][72];     // V^T chunk [d, s], 8-elem row pad
  __shared__ __bf16 sP[4][16][72];    // per-wave P staging

  const int wave = threadIdx.x >> 5;
  const int lane = threadIdx.x & 31;
  const int bh = blockIdx.y;
  const int b = bh >> 5, h = bh & 31, g = h >> 2;
  const int qt = blockIdx.x;
  const int qrow0 = qt * 64 + wave * 16;
  const int frow = lane & 15;
  const int halfadd = (lane & 16) ? 8 : 0;

  const __bf16* qbase  = qb + (((size_t)bh) * TQ + qrow0) * HD;
  const __bf16* kbase  = kb + ((size_t)(b * NG + g)) * STOT * HD;
  const __bf16* vtbase = vbt + ((size_t)(b * NG + g)) * HD * STOT;

  v16bf aq[4];
#pragma unroll
  for (int kt = 0; kt < 4; ++kt) aq[kt] = ldfragA(qbase + kt * 32, HD);

  v8f ctxa[8];
#pragma unroll
  for (int n = 0; n < 8; ++n) ctxa[n] = v8f{};
  float mr[8], lr[8];
#pragma unroll
  for (int r = 0; r < 8; ++r) { mr[r] = -1e30f; lr[r] = 0.0f; }

  const float scale = 0.08838834764831845f;   // 1/sqrt(128)
  const int qpos_base = SP + qrow0 + halfadd;
  const int nchunks = 17 + qt;

  for (int c = 0; c < nchunks; ++c) {
    const int s0 = c << 6;
#if defined(HAVE_TDM)
    if (wave == 0) {
      // K chunk: 64 rows x 128 elems, row pad 4 DWORDs per 64 DWORDs
      tdm_load_2d((unsigned)(size_t)&sK[0][0], kbase + (size_t)s0 * HD,
                  /*t_d0=*/HD, /*t_d1=*/STOT, /*tile_d0=*/HD, /*tile_d1=*/64,
                  /*stride=*/HD, /*interval=*/5, /*pad-1=*/3);
      // V^T chunk: 128 rows x 64 elems, row pad 4 DWORDs per 32 DWORDs
      tdm_load_2d((unsigned)(size_t)&sVT[0][0], vtbase + s0,
                  /*t_d0=*/STOT, /*t_d1=*/HD, /*tile_d0=*/64, /*tile_d1=*/HD,
                  /*stride=*/STOT, /*interval=*/4, /*pad-1=*/3);
      tdm_wait0();
    }
#else
    {   // stage K chunk (64x128)
      int row = threadIdx.x >> 1, half = (threadIdx.x & 1) * 64;
      const __bf16* src = kbase + (size_t)(s0 + row) * HD + half;
#pragma unroll
      for (int i = 0; i < 8; ++i)
        *(bf16x8*)&sK[row][half + i * 8] = *(const bf16x8*)(src + i * 8);
    }
    {   // stage V^T chunk (128x64)
      int row = threadIdx.x;
      const __bf16* src = vtbase + (size_t)row * STOT + s0;
#pragma unroll
      for (int i = 0; i < 8; ++i)
        *(bf16x8*)&sVT[row][i * 8] = *(const bf16x8*)(src + i * 8);
    }
#endif
    __syncthreads();

    // scores: 16x64 per wave
    v8f st[4];
#pragma unroll
    for (int j = 0; j < 4; ++j) st[j] = v8f{};
#pragma unroll
    for (int kt = 0; kt < 4; ++kt)
#pragma unroll
      for (int j = 0; j < 4; ++j)
        st[j] = wmma_bf16(aq[kt], ldfragB(&sK[j * 16][kt * 32], 136), st[j]);

    // online softmax per row
#pragma unroll
    for (int r = 0; r < 8; ++r) {
      const int qpos = qpos_base + r;
      float mx = -1e30f;
#pragma unroll
      for (int j = 0; j < 4; ++j) {
        int s = s0 + j * 16 + frow;
        float v = st[j][r] * scale;
        v = (s > qpos) ? -1e30f : v;
        st[j][r] = v;
        mx = fmaxf(mx, v);
      }
      mx = fmaxf(mx, __shfl_xor(mx, 1, 32));
      mx = fmaxf(mx, __shfl_xor(mx, 2, 32));
      mx = fmaxf(mx, __shfl_xor(mx, 4, 32));
      mx = fmaxf(mx, __shfl_xor(mx, 8, 32));
      float mnew = fmaxf(mr[r], mx);
      float corr = __expf(mr[r] - mnew);
      float rs = 0.0f;
#pragma unroll
      for (int j = 0; j < 4; ++j) {
        float p = __expf(st[j][r] - mnew);
        st[j][r] = p;
        rs += p;
      }
      rs += __shfl_xor(rs, 1, 32);
      rs += __shfl_xor(rs, 2, 32);
      rs += __shfl_xor(rs, 4, 32);
      rs += __shfl_xor(rs, 8, 32);
      lr[r] = lr[r] * corr + rs;
      mr[r] = mnew;
#pragma unroll
      for (int n = 0; n < 8; ++n) ctxa[n][r] *= corr;
#pragma unroll
      for (int j = 0; j < 4; ++j)
        sP[wave][halfadd + r][j * 16 + frow] = (__bf16)st[j][r];
    }

    // ctx += P @ V
#pragma unroll
    for (int kt2 = 0; kt2 < 2; ++kt2) {
      v16bf ap = ldfragA(&sP[wave][0][kt2 * 32], 72);
#pragma unroll
      for (int n = 0; n < 8; ++n)
        ctxa[n] = wmma_bf16(ap, ldfragB(&sVT[n * 16][kt2 * 32], 72), ctxa[n]);
    }
    __syncthreads();
  }

#pragma unroll
  for (int r = 0; r < 8; ++r) {
    float inv = 1.0f / lr[r];
    int t = qrow0 + halfadd + r;
    __bf16* dst = ctxb + ((size_t)(b * TQ + t)) * (NH * HD) + h * HD + frow;
#pragma unroll
    for (int n = 0; n < 8; ++n) dst[n * 16] = (__bf16)(ctxa[n][r] * inv);
  }
}

// ---------------------------------------------------------------------------
extern "C" void kernel_launch(void* const* d_in, const int* in_sizes, int n_in,
                              void* d_out, int out_size, void* d_ws, size_t ws_size,
                              hipStream_t stream) {
  const float* x      = (const float*)d_in[0];
  const float* Wq     = (const float*)d_in[1];
  const float* Wk     = (const float*)d_in[2];
  const float* Wv     = (const float*)d_in[3];
  const float* Wo     = (const float*)d_in[4];
  const float* qnw    = (const float*)d_in[5];
  const float* knw    = (const float*)d_in[6];
  const float* cosT   = (const float*)d_in[7];
  const float* sinT   = (const float*)d_in[8];
  const float* prev_k = (const float*)d_in[9];
  const float* prev_v = (const float*)d_in[10];

  float* out  = (float*)d_out;
  float* kall = out + (size_t)Bb * TQ * DIN;
  float* vall = kall + (size_t)Bb * NG * STOT * HD;

  char* ws = (char*)d_ws;
  size_t off = 0;
  auto carve = [&](size_t bytes) {
    void* p = ws + off;
    off += (bytes + 255) & ~(size_t)255;
    return p;
  };
  __bf16* xb    = (__bf16*)carve((size_t)Bb * TQ * DIN * 2);
  __bf16* wqb   = (__bf16*)carve((size_t)NH * HD * DIN * 2);
  __bf16* wkvb  = (__bf16*)carve((size_t)2 * NG * HD * DIN * 2);  // Wk ++ Wv
  __bf16* wob   = (__bf16*)carve((size_t)DIN * NH * HD * 2);
  float*  qraw  = (float*)carve((size_t)Bb * TQ * NH * HD * 4);
  float*  kvraw = (float*)carve((size_t)Bb * TQ * 2 * NG * HD * 4);
  __bf16* qbb   = (__bf16*)carve((size_t)Bb * NH * TQ * HD * 2);
  __bf16* kbb   = (__bf16*)carve((size_t)Bb * NG * STOT * HD * 2);
  __bf16* vbt   = (__bf16*)carve((size_t)Bb * NG * HD * STOT * 2);
  __bf16* ctxb  = (__bf16*)carve((size_t)Bb * TQ * NH * HD * 2);

  auto cast = [&](const float* src, __bf16* dst, size_t n) {
    int n4 = (int)(n / 4);
    cast_bf16_kernel<<<(n4 + 255) / 256, 256, 0, stream>>>(src, dst, n4);
  };
  cast(x, xb, (size_t)Bb * TQ * DIN);
  cast(Wq, wqb, (size_t)NH * HD * DIN);
  cast(Wk, wkvb, (size_t)NG * HD * DIN);
  cast(Wv, wkvb + (size_t)NG * HD * DIN, (size_t)NG * HD * DIN);
  cast(Wo, wob, (size_t)DIN * NH * HD);

  const int M = Bb * TQ;   // 2048
  gemm_bf16_nt_kernel<DIN, DIN><<<dim3(DIN / 128, M / 128), 256, 0, stream>>>(
      xb, wqb, qraw);                                    // Q: N=4096
  gemm_bf16_nt_kernel<2048, DIN><<<dim3(2048 / 128, M / 128), 256, 0, stream>>>(
      xb, wkvb, kvraw);                                  // K|V: N=2048

  {
    int n = Bb * NG * SP * HD;
    copyprev_kernel<<<(n + 255) / 256, 256, 0, stream>>>(prev_k, prev_v, kall,
                                                         vall, kbb, vbt);
  }
  postq_kernel<<<(Bb * TQ * NH) / 8, 256, 0, stream>>>(qraw, qnw, cosT, sinT, qbb);
  postk_kernel<<<(Bb * TQ * NG) / 8, 256, 0, stream>>>(kvraw, knw, cosT, sinT,
                                                       kall, kbb);
  postv_kernel<<<(Bb * TQ * NG) / 8, 256, 0, stream>>>(kvraw, vall, vbt);

  attn_kernel<<<dim3(TQ / 64, Bb * NH), 128, 0, stream>>>(qbb, kbb, vbt, ctxb);

  gemm_bf16_nt_kernel<DIN, NH * HD><<<dim3(DIN / 128, M / 128), 256, 0, stream>>>(
      ctxb, wob, out);                                   // out: N=4096

  (void)in_sizes; (void)n_in; (void)out_size; (void)ws_size;
}
